// MultiScaleBlock_10273561772250
// MI455X (gfx1250) — compile-verified, matrix operand order checked by reading, not run on hardware
//
#include <hip/hip_runtime.h>
#include <hip/hip_bf16.h>

// ---------- CDNA5 WMMA / TDM vector types ----------
typedef __attribute__((ext_vector_type(16))) __bf16 v16bf;
typedef __attribute__((ext_vector_type(8)))  float  v8f;
typedef __attribute__((ext_vector_type(4)))  unsigned int v4u;
typedef __attribute__((ext_vector_type(8)))  int    v8i;
typedef __attribute__((ext_vector_type(4)))  int    v4i;

#if defined(__has_builtin)
#  if __has_builtin(__builtin_amdgcn_tensor_load_to_lds)
#    define HAVE_TDM 1
#  endif
#endif
#ifndef HAVE_TDM
#  define HAVE_TDM 0
#endif

// ---------- problem dims (from setup_inputs) ----------
constexpr int B_    = 16;
constexpr int T_    = 8, Hh_ = 28, Ww_ = 28;
constexpr int D_    = 192;
constexpr int NH_   = 2;
constexpr int HD_   = 96;
constexpr int HID_  = 768;
constexpr int DOUT_ = 384;
constexpr int N_    = 1 + T_*Hh_*Ww_;      // 6273
constexpr int Tn_   = 8, Hn_ = 14, Wn_ = 14;
constexpr int Lq_   = 1 + Tn_*Hn_*Wn_;     // 1569
constexpr int LqPad_= 1600;                // Lq padded to multiple of 32 (zero-filled)
constexpr int BN_   = B_*N_;               // 100368 (multiple of 16)
constexpr int BLq_  = B_*Lq_;              // 25104  (multiple of 16)
constexpr int BH_   = B_*NH_;              // 32

// ---------- helpers ----------
__device__ __forceinline__ __bf16 f2bf(float f) {
    unsigned int u = __builtin_bit_cast(unsigned int, f);
    unsigned int r = u + 0x7fffu + ((u >> 16) & 1u);
    unsigned short h = (unsigned short)(r >> 16);
    return __builtin_bit_cast(__bf16, h);
}
// ISA 16-bit A-matrix 16x32 layout: lane group g=lane/16.
// elements 0..7 -> K = g*8 + i, elements 8..15 -> K = 16 + g*8 + (i-8)
__device__ __forceinline__ int kmap(int lane, int i) {
    int g = (lane >> 4) & 1;
    return (i < 8) ? (g*8 + i) : (16 + g*8 + (i - 8));
}

// ---------- zero fill (uint4 stores) ----------
__global__ void fill_zero_kernel(uint4* __restrict__ p, int nchunks) {
    int i = blockIdx.x * blockDim.x + threadIdx.x;
    if (i < nchunks) p[i] = make_uint4(0u, 0u, 0u, 0u);
}

// ---------- pack fp32 weights (KxN) into per-lane WMMA B fragments ----------
// out element idx = ((tn*(Kd/32)+kt)*32 + lane)*16 + i
//               <- W[(kt*32 + kmap(lane,i))*Nd + tn*16 + (lane&15)]
__global__ void pack_w_kernel(const float* __restrict__ W, __bf16* __restrict__ out,
                              int Kd, int Nd) {
    int idx = blockIdx.x * blockDim.x + threadIdx.x;
    if (idx >= Kd * Nd) return;
    int i    = idx & 15;
    int lane = (idx >> 4) & 31;
    int rest = idx >> 9;
    int kt   = rest % (Kd / 32);
    int tn   = rest / (Kd / 32);
    int kk   = kt * 32 + kmap(lane, i);
    int col  = tn * 16 + (lane & 15);
    out[idx] = f2bf(W[(size_t)kk * Nd + col]);
}

// ---------- LayerNorm: one wave per row; 3 output layouts ----------
// mode 0: out[row*C + c]
// mode 1: out[(bh*LqPad + tok)*C + c]          (padded token rows; bh=row/Lq, tok=row%Lq)
// mode 2: out[(bh*C + c)*LqPad + tok]          (transposed, padded token cols)
__global__ __launch_bounds__(256)
void ln_kernel(const float* __restrict__ x, const float* __restrict__ gm,
               const float* __restrict__ bt, __bf16* __restrict__ outH,
               float* __restrict__ outF, int rows, int C, int mode) {
    int row = blockIdx.x * blockDim.y + threadIdx.y;
    if (row >= rows) return;
    int lane = threadIdx.x;
    const float* xr = x + (size_t)row * C;
    float s = 0.f;
    for (int c = lane; c < C; c += 32) s += xr[c];
    for (int off = 16; off > 0; off >>= 1) s += __shfl_xor(s, off, 32);
    float mean = s / (float)C;
    float v = 0.f;
    for (int c = lane; c < C; c += 32) { float d = xr[c] - mean; v += d * d; }
    for (int off = 16; off > 0; off >>= 1) v += __shfl_xor(v, off, 32);
    float rstd = rsqrtf(v / (float)C + 1e-5f);
    int bh = row / Lq_, tok = row - bh * Lq_;   // only used for modes 1/2
    for (int c = lane; c < C; c += 32) {
        float y = (xr[c] - mean) * rstd * gm[c] + bt[c];
        if (mode == 0) {
            if (outH) outH[(size_t)row * C + c] = f2bf(y);
            if (outF) outF[(size_t)row * C + c] = y;
        } else if (mode == 1) {
            outH[((size_t)bh * LqPad_ + tok) * C + c] = f2bf(y);
        } else {
            outH[((size_t)bh * C + c) * LqPad_ + tok] = f2bf(y);
        }
    }
}

// ---------- GEMM: one block per 16-row M tile; A panel staged in LDS (TDM), ----------
// ---------- 4 waves split the N tiles; B read as packed v16bf fragments     ----------
__global__ __launch_bounds__(128)
void gemm_bf16_kernel(const __bf16* __restrict__ A, const v16bf* __restrict__ Wp,
                      const float* __restrict__ bias, const float* __restrict__ add,
                      float* __restrict__ outF, __bf16* __restrict__ outH,
                      int Kd, int Nd, int gelu) {
    extern __shared__ __bf16 Apanel[];          // 16 * Kd bf16
    const int tileM = blockIdx.x;
    const int lane  = threadIdx.x;
    const int wv    = threadIdx.y;
    const int l16   = lane & 15;
    const int g     = lane >> 4;

#if HAVE_TDM
    if (wv == 0) {
        // 2D tensor = the contiguous 16 x Kd panel itself (no OOB possible)
        unsigned long long gaddr = (unsigned long long)(const void*)(A + (size_t)tileM * 16 * Kd);
        unsigned int lds_off = (unsigned int)(unsigned long long)(void*)Apanel;
        unsigned int dim0 = (unsigned int)Kd;       // tensor_dim0 (elements)
        unsigned int dim1 = 16u;                    // tensor_dim1 (rows)
        unsigned long long st0 = (unsigned long long)Kd; // tensor_dim0_stride
        v4u g0;
        g0[0] = 1u;                                                  // count=1, user desc
        g0[1] = lds_off;                                             // lds_addr
        g0[2] = (unsigned int)(gaddr & 0xffffffffull);               // global_addr[31:0]
        g0[3] = (unsigned int)((gaddr >> 32) & 0x1ffffffull) | (2u << 30); // addr[56:32], type=2
        v8i g1;
        g1[0] = (int)(1u << 16);                                     // data_size=1 (2 bytes)
        g1[1] = (int)((dim0 & 0xffffu) << 16);                       // tensor_dim0[15:0]
        g1[2] = (int)((dim0 >> 16) | ((dim1 & 0xffffu) << 16));      // dim0[31:16] | dim1[15:0]
        g1[3] = (int)((dim1 >> 16) | (((unsigned)Kd & 0xffffu) << 16)); // dim1[31:16] | tile_dim0
        g1[4] = (int)16u;                                            // tile_dim1=16, tile_dim2=0
        g1[5] = (int)(unsigned int)(st0 & 0xffffffffull);            // stride0[31:0]
        g1[6] = (int)(unsigned int)((st0 >> 32) & 0xffffull);        // stride0[47:32] | d1stride[15:0]=0
        g1[7] = 0;
        v4i g2 = {0, 0, 0, 0};
        v4i g3 = {0, 0, 0, 0};
#if defined(__clang_major__) && (__clang_major__ >= 23)
        v8i g4 = {0, 0, 0, 0, 0, 0, 0, 0};
        __builtin_amdgcn_tensor_load_to_lds(g0, g1, g2, g3, g4, 0);
#else
        __builtin_amdgcn_tensor_load_to_lds(g0, g1, g2, g3, 0);
#endif
        __builtin_amdgcn_s_wait_tensorcnt(0);
    }
#else
    {   // cooperative copy fallback: panel is a contiguous 32*Kd-byte block
        const uint4* src = (const uint4*)(A + (size_t)tileM * 16 * Kd);
        uint4* dst = (uint4*)Apanel;
        int nchunk = (16 * Kd) / 8;
        for (int i = wv * 32 + lane; i < nchunk; i += 128) dst[i] = src[i];
    }
#endif
    __syncthreads();

    const int tilesN = Nd >> 4;
    const int ktiles = Kd >> 5;
    for (int tn = wv; tn < tilesN; tn += 4) {
        v8f acc = {0.f,0.f,0.f,0.f,0.f,0.f,0.f,0.f};
        for (int kt = 0; kt < ktiles; ++kt) {
            v16bf a, bw;
            #pragma unroll
            for (int i = 0; i < 16; ++i)
                a[i] = Apanel[l16 * Kd + kt * 32 + kmap(lane, i)];
            bw = Wp[((size_t)tn * ktiles + kt) * 32 + lane];
            acc = __builtin_amdgcn_wmma_f32_16x16x32_bf16(false, a, false, bw,
                                                          (short)0, acc, false, false);
        }
        int n = tn * 16 + l16;
        float bv = bias ? bias[n] : 0.f;
        #pragma unroll
        for (int r = 0; r < 8; ++r) {
            int m = tileM * 16 + r + 8 * g;
            float v = acc[r] + bv;
            if (add)  v += add[(size_t)m * Nd + n];
            if (gelu) v = 0.5f * v * (1.f + erff(v * 0.70710678f));
            if (outF) outF[(size_t)m * Nd + n] = v;
            if (outH) outH[(size_t)m * Nd + n] = f2bf(v);
        }
    }
}

// ---------- depthwise 3x3x3 conv pooling, stride (1,2,2), pad 1 ----------
__global__ void pool_kernel(const float* __restrict__ qkv, const float* __restrict__ w,
                            float* __restrict__ out, int which) {
    int idx = blockIdx.x * blockDim.x + threadIdx.x;
    if (idx >= B_ * NH_ * Lq_ * HD_) return;
    int c   = idx % HD_;
    int tok = (idx / HD_) % Lq_;
    int h   = (idx / (HD_ * Lq_)) % NH_;
    int b   = idx / (HD_ * Lq_ * NH_);
    int col = which * D_ + h * HD_ + c;
    if (tok == 0) { out[idx] = qkv[((size_t)b * N_) * (3 * D_) + col]; return; }
    int p  = tok - 1;
    int wn = p % Wn_;
    int hn = (p / Wn_) % Hn_;
    int tn = p / (Wn_ * Hn_);
    float s = 0.f;
    #pragma unroll
    for (int dt = 0; dt < 3; ++dt) {
        int t = tn + dt - 1;
        if (t < 0 || t >= T_) continue;
        #pragma unroll
        for (int dy = 0; dy < 3; ++dy) {
            int y = hn * 2 + dy - 1;
            if (y < 0 || y >= Hh_) continue;
            #pragma unroll
            for (int dx = 0; dx < 3; ++dx) {
                int xx = wn * 2 + dx - 1;
                if (xx < 0 || xx >= Ww_) continue;
                int n = 1 + (t * Hh_ + y) * Ww_ + xx;
                s += qkv[((size_t)b * N_ + n) * (3 * D_) + col] * w[c * 27 + dt * 9 + dy * 3 + dx];
            }
        }
    }
    out[idx] = s;
}

// ---------- maxpool (1,3,3) stride (1,2,2) skip path on x ----------
__global__ void maxpool_kernel(const float* __restrict__ x, float* __restrict__ out) {
    int idx = blockIdx.x * blockDim.x + threadIdx.x;
    if (idx >= BLq_ * D_) return;
    int c   = idx % D_;
    int tok = (idx / D_) % Lq_;
    int b   = idx / (D_ * Lq_);
    if (tok == 0) { out[idx] = x[((size_t)b * N_) * D_ + c]; return; }
    int p  = tok - 1;
    int wn = p % Wn_;
    int hn = (p / Wn_) % Hn_;
    int tn = p / (Wn_ * Hn_);
    float m = -3.0e38f;
    #pragma unroll
    for (int dy = 0; dy < 3; ++dy) {
        int y = hn * 2 + dy - 1;
        if (y < 0 || y >= Hh_) continue;
        #pragma unroll
        for (int dx = 0; dx < 3; ++dx) {
            int xx = wn * 2 + dx - 1;
            if (xx < 0 || xx >= Ww_) continue;
            int n = 1 + (tn * Hh_ + y) * Ww_ + xx;
            m = fmaxf(m, x[((size_t)b * N_ + n) * D_ + c]);
        }
    }
    out[idx] = m;
}

// ---------- flash attention: one wave per (bh, 16-row q tile) ----------
// Q,K: (BH, LqPad, HD) bf16 zero-padded rows; V: (BH, HD, LqPad) bf16 transposed, zero-padded cols.
__global__ __launch_bounds__(32)
void attn_kernel(const __bf16* __restrict__ Qm, const __bf16* __restrict__ Km,
                 const __bf16* __restrict__ Vm, __bf16* __restrict__ Om) {
    __shared__ float Pbuf[16 * 32];
    const int lane = threadIdx.x;
    const int l16  = lane & 15;
    const int g    = lane >> 4;
    const int qt   = blockIdx.x;
    const int bh   = blockIdx.y;
    const int b    = bh / NH_;
    const int h    = bh % NH_;
    const __bf16* Q  = Qm + (size_t)bh * LqPad_ * HD_;
    const __bf16* K  = Km + (size_t)bh * LqPad_ * HD_;
    const __bf16* Vt = Vm + (size_t)bh * HD_ * LqPad_;
    const float scale = 0.1020620726f;  // 96^-0.5

    const int qrow = qt * 16 + l16;     // < LqPad, padded rows are zero
    v16bf aq[3];
    #pragma unroll
    for (int t = 0; t < 3; ++t)
        #pragma unroll
        for (int i = 0; i < 16; ++i)
            aq[t][i] = Q[(size_t)qrow * HD_ + t * 32 + kmap(lane, i)];

    const v8f vzero = {0.f,0.f,0.f,0.f,0.f,0.f,0.f,0.f};
    v8f acc[6];
    #pragma unroll
    for (int t = 0; t < 6; ++t) acc[t] = vzero;
    float mrow[8], lrow[8];
    #pragma unroll
    for (int r = 0; r < 8; ++r) { mrow[r] = -3.0e38f; lrow[r] = 0.f; }

    for (int j0 = 0; j0 < Lq_; j0 += 32) {
        __builtin_prefetch(&K[(size_t)(j0 + 32) * HD_], 0, 1);
        const int tok0 = j0 + l16, tok1 = j0 + 16 + l16;   // < LqPad always
        v8f s0 = vzero, s1 = vzero;
        #pragma unroll
        for (int t = 0; t < 3; ++t) {
            v16bf bk0, bk1;
            #pragma unroll
            for (int i = 0; i < 16; ++i) {
                int kk = t * 32 + kmap(lane, i);
                bk0[i] = K[(size_t)tok0 * HD_ + kk];
                bk1[i] = K[(size_t)tok1 * HD_ + kk];
            }
            s0 = __builtin_amdgcn_wmma_f32_16x16x32_bf16(false, aq[t], false, bk0, (short)0, s0, false, false);
            s1 = __builtin_amdgcn_wmma_f32_16x16x32_bf16(false, aq[t], false, bk1, (short)0, s1, false, false);
        }
        float p0[8], p1[8], cmax[8], csum[8];
        #pragma unroll
        for (int r = 0; r < 8; ++r) {
            p0[r] = (tok0 < Lq_) ? s0[r] * scale : -3.0e38f;
            p1[r] = (tok1 < Lq_) ? s1[r] * scale : -3.0e38f;
            cmax[r] = fmaxf(p0[r], p1[r]);
        }
        #pragma unroll
        for (int off = 1; off < 16; off <<= 1)
            #pragma unroll
            for (int r = 0; r < 8; ++r)
                cmax[r] = fmaxf(cmax[r], __shfl_xor(cmax[r], off, 16));
        #pragma unroll
        for (int r = 0; r < 8; ++r) {
            float mnew = fmaxf(mrow[r], cmax[r]);
            float corr = __expf(mrow[r] - mnew);
            mrow[r] = mnew;
            p0[r] = __expf(p0[r] - mnew);
            p1[r] = __expf(p1[r] - mnew);
            csum[r] = p0[r] + p1[r];
            lrow[r] *= corr;
            #pragma unroll
            for (int t = 0; t < 6; ++t) acc[t][r] = acc[t][r] * corr;
        }
        #pragma unroll
        for (int off = 1; off < 16; off <<= 1)
            #pragma unroll
            for (int r = 0; r < 8; ++r)
                csum[r] += __shfl_xor(csum[r], off, 16);
        #pragma unroll
        for (int r = 0; r < 8; ++r) lrow[r] += csum[r];
        // stage P (C-layout -> LDS) and reread in A-layout as bf16
        #pragma unroll
        for (int r = 0; r < 8; ++r) {
            Pbuf[(r + 8 * g) * 32 + l16]      = p0[r];
            Pbuf[(r + 8 * g) * 32 + 16 + l16] = p1[r];
        }
        __syncthreads();
        v16bf ap;
        #pragma unroll
        for (int i = 0; i < 16; ++i)
            ap[i] = f2bf(Pbuf[l16 * 32 + kmap(lane, i)]);
        #pragma unroll
        for (int t = 0; t < 6; ++t) {
            v16bf bv;
            #pragma unroll
            for (int i = 0; i < 16; ++i)
                bv[i] = Vt[(size_t)(t * 16 + l16) * LqPad_ + j0 + kmap(lane, i)];
            acc[t] = __builtin_amdgcn_wmma_f32_16x16x32_bf16(false, ap, false, bv, (short)0, acc[t], false, false);
        }
        __syncthreads();
    }
    #pragma unroll
    for (int r = 0; r < 8; ++r) {
        int m = qt * 16 + r + 8 * g;
        if (m < Lq_) {
            float inv = 1.f / lrow[r];
            #pragma unroll
            for (int t = 0; t < 6; ++t)
                Om[((size_t)(b * Lq_ + m)) * D_ + h * HD_ + t * 16 + l16] = f2bf(acc[t][r] * inv);
        }
    }
}

// ---------- host launcher ----------
extern "C" void kernel_launch(void* const* d_in, const int* in_sizes, int n_in,
                              void* d_out, int out_size, void* d_ws, size_t ws_size,
                              hipStream_t stream) {
    (void)in_sizes; (void)n_in; (void)out_size; (void)ws_size;
    const float* x      = (const float*)d_in[0];
    const float* ln1_g  = (const float*)d_in[1];
    const float* ln1_b  = (const float*)d_in[2];
    const float* qkv_w  = (const float*)d_in[3];
    const float* qkv_b  = (const float*)d_in[4];
    const float* pq_w   = (const float*)d_in[5];
    const float* pk_w   = (const float*)d_in[6];
    const float* pv_w   = (const float*)d_in[7];
    const float* nq_g   = (const float*)d_in[8];
    const float* nq_b   = (const float*)d_in[9];
    const float* nk_g   = (const float*)d_in[10];
    const float* nk_b   = (const float*)d_in[11];
    const float* nv_g   = (const float*)d_in[12];
    const float* nv_b   = (const float*)d_in[13];
    const float* proj_w = (const float*)d_in[14];
    const float* proj_b = (const float*)d_in[15];
    const float* ln2_g  = (const float*)d_in[16];
    const float* ln2_b  = (const float*)d_in[17];
    const float* fc1_w  = (const float*)d_in[18];
    const float* fc1_b  = (const float*)d_in[19];
    const float* fc2_w  = (const float*)d_in[20];
    const float* fc2_b  = (const float*)d_in[21];
    const float* projd_w= (const float*)d_in[22];
    const float* projd_b= (const float*)d_in[23];

    char* ws = (char*)d_ws;
    size_t off = 0;
    auto alloc = [&](size_t bytes) -> char* {
        char* p = ws + off;
        off += (bytes + 255) & ~(size_t)255;
        return p;
    };
    __bf16* qkvw_p   = (__bf16*)alloc((size_t)D_ * 3 * D_ * 2);
    __bf16* projw_p  = (__bf16*)alloc((size_t)D_ * D_ * 2);
    __bf16* fc1w_p   = (__bf16*)alloc((size_t)D_ * HID_ * 2);
    __bf16* fc2w_p   = (__bf16*)alloc((size_t)HID_ * DOUT_ * 2);
    __bf16* projdw_p = (__bf16*)alloc((size_t)D_ * DOUT_ * 2);
    __bf16* xn_h     = (__bf16*)alloc((size_t)BN_ * D_ * 2);
    float*  qkv_f    = (float*) alloc((size_t)BN_ * 3 * D_ * 4);
    size_t pool_elems = (size_t)BH_ * Lq_ * HD_;
    size_t pad_elems  = (size_t)BH_ * LqPad_ * HD_;
    float*  pq_f = (float*)alloc(pool_elems * 4);
    float*  pk_f = (float*)alloc(pool_elems * 4);
    float*  pv_f = (float*)alloc(pool_elems * 4);
    __bf16* q_h  = (__bf16*)alloc(pad_elems * 2);   // (BH, LqPad, HD)
    __bf16* k_h  = (__bf16*)alloc(pad_elems * 2);   // (BH, LqPad, HD)
    __bf16* v_t  = (__bf16*)alloc(pad_elems * 2);   // (BH, HD, LqPad)
    __bf16* o_h  = (__bf16*)alloc((size_t)BLq_ * D_ * 2);
    float*  xres_f = (float*)alloc((size_t)BLq_ * D_ * 4);
    float*  x1_f   = (float*)alloc((size_t)BLq_ * D_ * 4);
    __bf16* xn2_h  = (__bf16*)alloc((size_t)BLq_ * D_ * 2);
    __bf16* hmid_h = (__bf16*)alloc((size_t)BLq_ * HID_ * 2);
    float*  tmp_f  = (float*)alloc((size_t)BLq_ * DOUT_ * 4);

    auto cdiv = [](long long a, long long b) { return (int)((a + b - 1) / b); };

    // pack weights into WMMA B-fragment layout (bf16)
    pack_w_kernel<<<cdiv(D_*3*D_, 256), 256, 0, stream>>>(qkv_w,   qkvw_p,   D_, 3*D_);
    pack_w_kernel<<<cdiv(D_*D_,   256), 256, 0, stream>>>(proj_w,  projw_p,  D_, D_);
    pack_w_kernel<<<cdiv(D_*HID_, 256), 256, 0, stream>>>(fc1_w,   fc1w_p,   D_, HID_);
    pack_w_kernel<<<cdiv(HID_*DOUT_,256),256, 0, stream>>>(fc2_w,   fc2w_p,   HID_, DOUT_);
    pack_w_kernel<<<cdiv(D_*DOUT_,256),  256, 0, stream>>>(projd_w, projdw_p, D_, DOUT_);

    // zero padded attention operand buffers
    {
        int nchunks = (int)(pad_elems * 2 / 16);
        fill_zero_kernel<<<cdiv(nchunks,256), 256, 0, stream>>>((uint4*)q_h, nchunks);
        fill_zero_kernel<<<cdiv(nchunks,256), 256, 0, stream>>>((uint4*)k_h, nchunks);
        fill_zero_kernel<<<cdiv(nchunks,256), 256, 0, stream>>>((uint4*)v_t, nchunks);
    }

    dim3 lnblk(32, 8);
    dim3 gmblk(32, 4);

    // 1) LN1 -> xn (bf16)
    ln_kernel<<<cdiv(BN_, 8), lnblk, 0, stream>>>(x, ln1_g, ln1_b, xn_h, nullptr, BN_, D_, 0);
    // 2) QKV GEMM: (BN x 192) @ (192 x 576) + b -> f32   [A panel via TDM -> LDS]
    gemm_bf16_kernel<<<BN_/16, gmblk, 16*D_*2, stream>>>(
        xn_h, (const v16bf*)qkvw_p, qkv_b, nullptr, qkv_f, nullptr, D_, 3*D_, 0);
    // 3) depthwise pooling of q/k/v
    int pn = (int)pool_elems;
    pool_kernel<<<cdiv(pn, 256), 256, 0, stream>>>(qkv_f, pq_w, pq_f, 0);
    pool_kernel<<<cdiv(pn, 256), 256, 0, stream>>>(qkv_f, pk_w, pk_f, 1);
    pool_kernel<<<cdiv(pn, 256), 256, 0, stream>>>(qkv_f, pv_w, pv_f, 2);
    // 4) per-head LN -> padded bf16 (q,k row-padded; v transposed)
    int prows = BH_ * Lq_;
    ln_kernel<<<cdiv(prows, 8), lnblk, 0, stream>>>(pq_f, nq_g, nq_b, q_h, nullptr, prows, HD_, 1);
    ln_kernel<<<cdiv(prows, 8), lnblk, 0, stream>>>(pk_f, nk_g, nk_b, k_h, nullptr, prows, HD_, 1);
    ln_kernel<<<cdiv(prows, 8), lnblk, 0, stream>>>(pv_f, nv_g, nv_b, v_t, nullptr, prows, HD_, 2);
    // 5) flash attention -> o (bf16, head-concat layout)
    {
        dim3 agrd((Lq_ + 15) / 16, BH_);
        attn_kernel<<<agrd, dim3(32), 0, stream>>>(q_h, k_h, v_t, o_h);
    }
    // 6) maxpool skip on x -> xres
    maxpool_kernel<<<cdiv((long long)BLq_ * D_, 256), 256, 0, stream>>>(x, xres_f);
    // 7) proj GEMM + residual add -> x1 (f32)
    gemm_bf16_kernel<<<BLq_/16, gmblk, 16*D_*2, stream>>>(
        o_h, (const v16bf*)projw_p, proj_b, xres_f, x1_f, nullptr, D_, D_, 0);
    // 8) LN2 -> xn2 (bf16)
    ln_kernel<<<cdiv(BLq_, 8), lnblk, 0, stream>>>(x1_f, ln2_g, ln2_b, xn2_h, nullptr, BLq_, D_, 0);
    // 9) fc1 + GELU -> hmid (bf16)
    gemm_bf16_kernel<<<BLq_/16, gmblk, 16*D_*2, stream>>>(
        xn2_h, (const v16bf*)fc1w_p, fc1_b, nullptr, nullptr, hmid_h, D_, HID_, 1);
    // 10) projd -> tmp (f32)
    gemm_bf16_kernel<<<BLq_/16, gmblk, 16*D_*2, stream>>>(
        xn2_h, (const v16bf*)projdw_p, projd_b, nullptr, tmp_f, nullptr, D_, DOUT_, 0);
    // 11) fc2 + bias + tmp -> out (f32)
    gemm_bf16_kernel<<<BLq_/16, gmblk, 16*HID_*2, stream>>>(
        hmid_h, (const v16bf*)fc2w_p, fc2_b, tmp_f, (float*)d_out, nullptr, HID_, DOUT_, 0);
}
